// MultiHeadAttention_69131793597008
// MI455X (gfx1250) — matrix operand order repaired and checked by reading, not compiled, not run on hardware
//
#include <hip/hip_runtime.h>
#include <hip/hip_bf16.h>

#define B_  4
#define T_  2048
#define C_  1024
#define H_  16
#define D_  64
#define N3_ 3072
#define NEG_INF (-__builtin_inff())

typedef _Float16 v16h __attribute__((ext_vector_type(16)));
typedef _Float16 v8h  __attribute__((ext_vector_type(8)));
typedef float    v8f  __attribute__((ext_vector_type(8)));

union V16 { v16h v; v8h h[2]; _Float16 e[16]; };

// Build a 16-half A-fragment from two contiguous 8-half (16B) groups,
// in contiguous k order (k = 16h + s, callers pass p1 = p0 + 8), same as the B fragments.
__device__ __forceinline__ v16h load_split(const _Float16* p0, const _Float16* p1) {
  V16 u;
  u.h[0] = *(const v8h*)p0;
  u.h[1] = *(const v8h*)p1;
  return u.v;
}

// Exchange with lane^16 partner (wave32) — softmax reductions and S^T->P^T
// fragment redistribution.
__device__ __forceinline__ float xor16(float v) {
  int lane = (int)(threadIdx.x & 31u);
  int i = __builtin_bit_cast(int, v);
  int j = __builtin_amdgcn_ds_bpermute((lane ^ 16) << 2, i);
  return __builtin_bit_cast(float, j);
}

__device__ __forceinline__ v8f wmma16(v16h a, v16h b, v8f c) {
  // D = A(16x32 f16) * B(32x16 f16) + C(16x16 f32)
  return __builtin_amdgcn_wmma_f32_16x16x32_f16(false, a, false, b, (short)0, c,
                                                false, false);
}

// 32x64 wave tile, K = C_ = 1024. k-loop unrolled by 2 with ping-pong
// fragment sets: no register-rotation copies, loads for one set overlap
// WMMAs of the other.
__device__ __forceinline__ void gemm_32x64(const _Float16* __restrict__ a0p,
                                           const _Float16* __restrict__ a1p,
                                           const _Float16* __restrict__ b0p,
                                           const _Float16* __restrict__ b1p,
                                           const _Float16* __restrict__ b2p,
                                           const _Float16* __restrict__ b3p,
                                           v8f acc[8]) {
  v16h a0A = load_split(a0p, a0p + 8);
  v16h a1A = load_split(a1p, a1p + 8);
  v16h b0A = *(const v16h*)b0p;
  v16h b1A = *(const v16h*)b1p;
  v16h b2A = *(const v16h*)b2p;
  v16h b3A = *(const v16h*)b3p;
  for (unsigned kc = 0; kc < C_ / 32; kc += 2) {
    unsigned kb1 = (kc + 1) * 32;
    v16h a0B = load_split(a0p + kb1, a0p + kb1 + 8);
    v16h a1B = load_split(a1p + kb1, a1p + kb1 + 8);
    v16h b0B = *(const v16h*)(b0p + kb1);
    v16h b1B = *(const v16h*)(b1p + kb1);
    v16h b2B = *(const v16h*)(b2p + kb1);
    v16h b3B = *(const v16h*)(b3p + kb1);
    acc[0] = wmma16(a0A, b0A, acc[0]);
    acc[1] = wmma16(a0A, b1A, acc[1]);
    acc[2] = wmma16(a0A, b2A, acc[2]);
    acc[3] = wmma16(a0A, b3A, acc[3]);
    acc[4] = wmma16(a1A, b0A, acc[4]);
    acc[5] = wmma16(a1A, b1A, acc[5]);
    acc[6] = wmma16(a1A, b2A, acc[6]);
    acc[7] = wmma16(a1A, b3A, acc[7]);
    unsigned kb2 = (kc + 2 < C_ / 32 ? kc + 2 : kc) * 32;  // last pair: benign reload
    a0A = load_split(a0p + kb2, a0p + kb2 + 8);
    a1A = load_split(a1p + kb2, a1p + kb2 + 8);
    b0A = *(const v16h*)(b0p + kb2);
    b1A = *(const v16h*)(b1p + kb2);
    b2A = *(const v16h*)(b2p + kb2);
    b3A = *(const v16h*)(b3p + kb2);
    acc[0] = wmma16(a0B, b0B, acc[0]);
    acc[1] = wmma16(a0B, b1B, acc[1]);
    acc[2] = wmma16(a0B, b2B, acc[2]);
    acc[3] = wmma16(a0B, b3B, acc[3]);
    acc[4] = wmma16(a1B, b0B, acc[4]);
    acc[5] = wmma16(a1B, b1B, acc[5]);
    acc[6] = wmma16(a1B, b2B, acc[6]);
    acc[7] = wmma16(a1B, b3B, acc[7]);
  }
}

// Online-softmax update for one 16-query column set over a 32-key block.
// s0/s1 hold S^T (keys k0..k0+15 / k0+16..k0+31 as C-layout rows, query = lane
// column). Rescales o[], updates (m,lsum), returns the P^T B-fragment.
__device__ __forceinline__ v16h softmax_update(v8f s0, v8f s1, unsigned k0,
                                               unsigned q, bool diag, unsigned hi,
                                               float& m, float& lsum, v8f o[4]) {
  const float sc = 0.125f * 1.4426950408889634f;   // log2(e)/sqrt(D)
  float t0[8], t1[8], bmax = NEG_INF;
#pragma unroll
  for (unsigned r = 0; r < 8; ++r) {
    unsigned key0 = k0 + r + hi * 8;
    unsigned key1 = key0 + 16;
    t0[r] = (diag && key0 > q) ? NEG_INF : s0[r] * sc;
    t1[r] = (diag && key1 > q) ? NEG_INF : s1[r] * sc;
    bmax = fmaxf(bmax, fmaxf(t0[r], t1[r]));
  }
  bmax = fmaxf(bmax, xor16(bmax));
  float mnew  = fmaxf(m, bmax);
  float alpha = __builtin_amdgcn_exp2f(m - mnew);
  m = mnew;
  lsum *= alpha;
#pragma unroll
  for (unsigned dc = 0; dc < 4; ++dc) o[dc] = o[dc] * alpha;
  float p0[8], p1[8], ls = 0.f;
#pragma unroll
  for (unsigned r = 0; r < 8; ++r) {
    p0[r] = __builtin_amdgcn_exp2f(t0[r] - mnew);
    p1[r] = __builtin_amdgcn_exp2f(t1[r] - mnew);
    ls += p0[r] + p1[r];
  }
  lsum += ls + xor16(ls);
  V16 pb;
#pragma unroll
  for (unsigned r = 0; r < 8; ++r) {
    float x0 = xor16(p0[r]);
    float x1 = xor16(p1[r]);
    pb.e[r]     = (hi == 0) ? (_Float16)p0[r] : (_Float16)x1;
    pb.e[8 + r] = (hi == 0) ? (_Float16)x0    : (_Float16)p1[r];
  }
  return pb.v;
}

// ---------------------------------------------------------------- conversions
__global__ __launch_bounds__(256) void cvt_f32_f16(const float* __restrict__ in,
                                                   _Float16* __restrict__ out,
                                                   unsigned n) {
  unsigned i = blockIdx.x * 256u + threadIdx.x;
  if (i < n) out[i] = (_Float16)in[i];
}

// wt[n][k] = (f16) w[k][n]  — contiguous-k B-fragment gathers
__global__ __launch_bounds__(256) void transpose_cvt(const float* __restrict__ w,
                                                     _Float16* __restrict__ wt,
                                                     unsigned K, unsigned N) {
  unsigned i = blockIdx.x * 256u + threadIdx.x;
  if (i < N * K) {
    unsigned n = i / K, k = i - n * K;
    wt[i] = (_Float16)w[(size_t)k * N + n];
  }
}

// ---------------------------------------------------------------- QKV GEMM
// x[8192,1024] @ w_attn + b_attn, scattered to q,k:[B,H,T,D], v:[B,H,D,T]
__global__ __launch_bounds__(256) void qkv_gemm(const _Float16* __restrict__ xh,
                                                const _Float16* __restrict__ wt,
                                                const float* __restrict__ bias,
                                                _Float16* __restrict__ qh,
                                                _Float16* __restrict__ kh,
                                                _Float16* __restrict__ vh) {
  const unsigned lane = threadIdx.x & 31u;
  const unsigned wv   = threadIdx.x >> 5;
  const unsigned lid  = lane & 15u, hi = lane >> 4;
  unsigned wg = blockIdx.x * 8u + wv;          // 12288 wave tiles (32m x 64n)
  unsigned tm = wg / (N3_ / 64);               // 256 row tiles of 32
  unsigned tn = (wg % (N3_ / 64)) * 64;        // 48 col groups of 64

  const _Float16* a0p = xh + (size_t)(tm * 32 + lid) * C_ + hi * 16;
  const _Float16* a1p = a0p + 16 * C_;
  const _Float16* b0p = wt + (size_t)(tn + lid) * C_ + hi * 16;
  v8f acc[8] = {};
  gemm_32x64(a0p, a1p, b0p, b0p + 16 * C_, b0p + 32 * C_, b0p + 48 * C_, acc);

#pragma unroll
  for (unsigned half = 0; half < 2; ++half) {
#pragma unroll
    for (unsigned j = 0; j < 4; ++j) {
      unsigned n = tn + 16 * j + lid;
      float bv = bias[n];
      unsigned which = n >> 10;                // n / C_
      unsigned c = n & (C_ - 1);
      unsigned h = c >> 6, d = c & (D_ - 1);
#pragma unroll
      for (unsigned r = 0; r < 8; ++r) {
        unsigned gm = tm * 32 + half * 16 + r + hi * 8;   // global row b*T+t
        unsigned b = gm >> 11, t = gm & (T_ - 1);
        _Float16 hv = (_Float16)(acc[half * 4 + j][r] + bv);
        if (which == 0)      qh[((size_t)(b * H_ + h) * T_ + t) * D_ + d] = hv;
        else if (which == 1) kh[((size_t)(b * H_ + h) * T_ + t) * D_ + d] = hv;
        else                 vh[((size_t)(b * H_ + h) * D_ + d) * T_ + t] = hv;
      }
    }
  }
}

// ---------------------------------------------------------------- attention
// One wave per (b,h,32-query block): two 16-query column sets share each K and
// V fragment (16 WMMAs per 16 b128 loads). S^T = K·Q^T, online softmax (each
// query column lives in lanes q and q+16 only), O^T = V^T·P^T. No LDS, no
// barriers, EXEC all-ones for every WMMA.
__global__ __launch_bounds__(256) void attn_kernel(const _Float16* __restrict__ qh,
                                                   const _Float16* __restrict__ kh,
                                                   const _Float16* __restrict__ vh,
                                                   _Float16* __restrict__ yh) {
  const unsigned lane = threadIdx.x & 31u;
  const unsigned wv   = threadIdx.x >> 5;
  const unsigned lid  = lane & 15u, hi = lane >> 4;
  unsigned wg = blockIdx.x * 8u + wv;          // 4096 waves
  unsigned qb = wg & (T_ / 32 - 1);            // 64 query blocks of 32
  unsigned h  = (wg >> 6) & (H_ - 1);
  unsigned b  = wg >> 10;
  unsigned q0 = qb * 32;
  size_t bh = (size_t)(b * H_ + h);
  const _Float16* Qb = qh + bh * T_ * D_;
  const _Float16* Kb = kh + bh * T_ * D_;
  const _Float16* Vb = vh + bh * D_ * T_;

  unsigned qA = q0 + lid, qB = q0 + 16 + lid;
  // Loop-invariant Q^T B-fragments, two query sets x two d chunks
  v16h qfA0 = *(const v16h*)(Qb + (size_t)qA * D_ + hi * 16);
  v16h qfA1 = *(const v16h*)(Qb + (size_t)qA * D_ + 32 + hi * 16);
  v16h qfB0 = *(const v16h*)(Qb + (size_t)qB * D_ + hi * 16);
  v16h qfB1 = *(const v16h*)(Qb + (size_t)qB * D_ + 32 + hi * 16);

  v8f oA[4] = {}, oB[4] = {};
  float mA = NEG_INF, lsA = 0.f, mB = NEG_INF, lsB = 0.f;
  unsigned nblk = qb + 1;                      // causal key blocks of 32
  for (unsigned kbi = 0; kbi < nblk; ++kbi) {
    unsigned k0 = kbi * 32;
    const _Float16* K0 = Kb + (size_t)(k0 + lid) * D_;  // keys k0..k0+15
    const _Float16* K1 = K0 + 16 * D_;                  // keys k0+16..k0+31
    // Issue 8 K-fragment loads, then 8 S-WMMAs (both query sets reuse them).
    v16h ka00 = load_split(K0 + hi * 16,      K0 + 8 + hi * 16);
    v16h ka01 = load_split(K0 + 32 + hi * 16, K0 + 40 + hi * 16);
    v16h ka10 = load_split(K1 + hi * 16,      K1 + 8 + hi * 16);
    v16h ka11 = load_split(K1 + 32 + hi * 16, K1 + 40 + hi * 16);
    if (kbi + 1 < nblk) {                      // prefetch next block's K rows
      __builtin_prefetch((const void*)(K0 + 32 * D_), 0, 1);
      __builtin_prefetch((const void*)(K1 + 32 * D_), 0, 1);
    }
    v8f sA0 = {}, sA1 = {}, sB0 = {}, sB1 = {};
    sA0 = wmma16(ka00, qfA0, sA0);
    sA0 = wmma16(ka01, qfA1, sA0);
    sA1 = wmma16(ka10, qfA0, sA1);
    sA1 = wmma16(ka11, qfA1, sA1);
    sB0 = wmma16(ka00, qfB0, sB0);
    sB0 = wmma16(ka01, qfB1, sB0);
    sB1 = wmma16(ka10, qfB0, sB1);
    sB1 = wmma16(ka11, qfB1, sB1);
    // V^T A-fragments issued before the softmax VALU work (overlap latency).
    v16h va[4];
#pragma unroll
    for (unsigned dc = 0; dc < 4; ++dc) {
      const _Float16* Vr = Vb + (size_t)(dc * 16 + lid) * T_ + k0 + hi * 16;
      va[dc] = load_split(Vr, Vr + 8);
    }
    bool diag = (kbi + 1 == nblk);             // only final block needs masking
    v16h pbA = softmax_update(sA0, sA1, k0, qA, diag, hi, mA, lsA, oA);
    v16h pbB = softmax_update(sB0, sB1, k0, qB, diag, hi, mB, lsB, oB);
#pragma unroll
    for (unsigned dc = 0; dc < 4; ++dc) {
      oA[dc] = wmma16(va[dc], pbA, oA[dc]);
      oB[dc] = wmma16(va[dc], pbB, oB[dc]);
    }
  }
  float invA = 1.f / lsA, invB = 1.f / lsB;
  size_t ybA = ((size_t)(b * T_ + qA) * H_ + h) * D_;  // y[B,T,H,D] == [B,T,C]
  size_t ybB = ((size_t)(b * T_ + qB) * H_ + h) * D_;
#pragma unroll
  for (unsigned dc = 0; dc < 4; ++dc) {
    v8h stA, stB;
#pragma unroll
    for (unsigned r = 0; r < 8; ++r) {
      stA[r] = (_Float16)(oA[dc][r] * invA);
      stB[r] = (_Float16)(oB[dc][r] * invB);
    }
    *(v8h*)(yh + ybA + dc * 16 + hi * 8) = stA;
    *(v8h*)(yh + ybB + dc * 16 + hi * 8) = stB;
  }
}

// ---------------------------------------------------------------- output proj
__global__ __launch_bounds__(256) void proj_gemm(const _Float16* __restrict__ yh,
                                                 const _Float16* __restrict__ wt,
                                                 const float* __restrict__ bias,
                                                 float* __restrict__ out) {
  const unsigned lane = threadIdx.x & 31u;
  const unsigned wv   = threadIdx.x >> 5;
  const unsigned lid  = lane & 15u, hi = lane >> 4;
  unsigned wg = blockIdx.x * 8u + wv;          // 4096 wave tiles (32m x 64n)
  unsigned tm = wg / (C_ / 64);
  unsigned tn = (wg % (C_ / 64)) * 64;
  const _Float16* a0p = yh + (size_t)(tm * 32 + lid) * C_ + hi * 16;
  const _Float16* a1p = a0p + 16 * C_;
  const _Float16* b0p = wt + (size_t)(tn + lid) * C_ + hi * 16;
  v8f acc[8] = {};
  gemm_32x64(a0p, a1p, b0p, b0p + 16 * C_, b0p + 32 * C_, b0p + 48 * C_, acc);

#pragma unroll
  for (unsigned half = 0; half < 2; ++half) {
#pragma unroll
    for (unsigned j = 0; j < 4; ++j) {
      unsigned n = tn + 16 * j + lid;
      float bv = bias[n];
#pragma unroll
      for (unsigned r = 0; r < 8; ++r) {
        unsigned gm = tm * 32 + half * 16 + r + hi * 8;
        out[(size_t)gm * C_ + n] = acc[half * 4 + j][r] + bv;
      }
    }
  }
}

// ---------------------------------------------------------------- launcher
extern "C" void kernel_launch(void* const* d_in, const int* in_sizes, int n_in,
                              void* d_out, int out_size, void* d_ws, size_t ws_size,
                              hipStream_t stream) {
  (void)in_sizes; (void)n_in; (void)out_size; (void)ws_size;
  const float* x      = (const float*)d_in[0];
  const float* w_attn = (const float*)d_in[1];
  const float* b_attn = (const float*)d_in[2];
  const float* w_proj = (const float*)d_in[3];
  const float* b_proj = (const float*)d_in[4];
  float* out = (float*)d_out;

  char* ws = (char*)d_ws;
  size_t off = 0;
  auto alloc = [&](size_t bytes) {
    void* p = ws + off;
    off += (bytes + 255) & ~(size_t)255;
    return p;
  };
  const size_t MT = (size_t)B_ * T_;                         // 8192 rows
  _Float16* xh  = (_Float16*)alloc(MT * C_ * 2);             // 16 MB
  _Float16* wat = (_Float16*)alloc((size_t)N3_ * C_ * 2);    //  6 MB
  _Float16* wpt = (_Float16*)alloc((size_t)C_ * C_ * 2);     //  2 MB
  _Float16* qh  = (_Float16*)alloc(MT * C_ * 2);             // 16 MB
  _Float16* kh  = (_Float16*)alloc(MT * C_ * 2);             // 16 MB
  _Float16* vh  = (_Float16*)alloc(MT * C_ * 2);             // 16 MB (transposed)
  _Float16* yh  = (_Float16*)alloc(MT * C_ * 2);             // 16 MB

  unsigned nx = (unsigned)(MT * C_);
  cvt_f32_f16<<<(nx + 255) / 256, 256, 0, stream>>>(x, xh, nx);
  unsigned nwa = N3_ * C_;
  transpose_cvt<<<(nwa + 255) / 256, 256, 0, stream>>>(w_attn, wat, C_, N3_);
  unsigned nwp = C_ * C_;
  transpose_cvt<<<(nwp + 255) / 256, 256, 0, stream>>>(w_proj, wpt, C_, C_);

  qkv_gemm<<<(256 * (N3_ / 64)) / 8, 256, 0, stream>>>(xh, wat, b_attn, qh, kh, vh);
  attn_kernel<<<(B_ * H_ * (T_ / 32)) / 8, 256, 0, stream>>>(qh, kh, vh, yh);
  proj_gemm<<<(256 * (C_ / 64)) / 8, 256, 0, stream>>>(yh, wpt, b_proj, out);
}
